// InferenceModel_33603824124428
// MI455X (gfx1250) — matrix operand order repaired and checked by reading, not compiled, not run on hardware
//
#include <hip/hip_runtime.h>
#include <hip/hip_bf16.h>

typedef __attribute__((ext_vector_type(16))) _Float16 v16h;
typedef __attribute__((ext_vector_type(8)))  _Float16 h8;
typedef __attribute__((ext_vector_type(8)))  float    v8f;

#define NBATCH   16
#define NROW     25200
#define NCH      85
#define NC       80
#define NBINS    2048
#define CAP      4096          // gather/sort capacity (power of 2)
#define NCAND    1000          // MAX_CAND
#define NPAD     1024          // padded candidate count (32 K-tiles of 32)
#define BSTRIDE  1032          // B panel row stride in halves (bank-skewed)
#define MAXDET   300
#define CONF     0.001f
#define IOU_T    0.6f
#define MAX_WH   4096.0f

// ---------------- workspace layout (bytes) ----------------
#define WS_HIST_OFF   0
#define WS_SUPW_OFF   ((size_t)NBATCH * NBINS * 4)
#define WS_MERG_OFF   (WS_SUPW_OFF + (size_t)NBATCH * NPAD * 32 * 4)

__global__ void nms_init(unsigned* __restrict__ hist) {
    int i = blockIdx.x * blockDim.x + threadIdx.x;
    if (i < NBATCH * NBINS) hist[i] = 0u;
}

// Pass 1: per-batch score histogram (streaming, memory-bound).
__global__ void nms_hist(const float* __restrict__ pred, unsigned* __restrict__ hist) {
    __shared__ unsigned lh[NBINS];
    const int b = blockIdx.x;
    for (int i = threadIdx.x; i < NBINS; i += blockDim.x) lh[i] = 0u;
    __syncthreads();
    const float* pb = pred + (size_t)b * NROW * NCH;
    const int nthr = blockDim.x * gridDim.y;
    const int t = blockIdx.y * blockDim.x + threadIdx.x;
    for (int r = t; r < NROW; r += nthr) {
        const float* p = pb + (size_t)r * NCH;
        __builtin_prefetch(p + (size_t)nthr * NCH, 0, 0);   // global_prefetch_b8
        float obj = p[4];
        if (obj > CONF) {
            #pragma unroll 4
            for (int c = 0; c < NC; ++c) {
                float s = p[5 + c] * obj;
                if (s > CONF) {
                    int bin = (int)(s * (float)NBINS);
                    bin = bin > NBINS - 1 ? NBINS - 1 : (bin < 0 ? 0 : bin);
                    atomicAdd(&lh[bin], 1u);
                }
            }
        }
    }
    __syncthreads();
    for (int i = threadIdx.x; i < NBINS; i += blockDim.x)
        if (lh[i]) atomicAdd(&hist[b * NBINS + i], lh[i]);
}

// Pass 2: one block per batch: gather -> sort -> IoU bitmaps -> greedy
// suppression -> WMMA merge matmul -> compaction.
__global__ void __launch_bounds__(256)
nms_main(const float* __restrict__ pred,
         const unsigned* __restrict__ hist,
         unsigned* __restrict__ supw_g,
         float* __restrict__ merged_g,
         float* __restrict__ out) {
    __shared__ union alignas(16) {
        struct { float sv[CAP]; int si[CAP]; } s;   // sort buffer (32 KB)
        _Float16 BhT[16][BSTRIDE];                  // transposed B panel (33 KB)
        int dest[NPAD];                             // output ranks
    } U;
    __shared__ float    bx[NPAD][4];     // xyxy boxes (no class offset)
    __shared__ float    clsA[NPAD];      // class id as float
    __shared__ float    topv[NPAD];      // top scores
    __shared__ alignas(16) _Float16 tv16[NPAD];  // valid ? (f16)score : 0
    __shared__ int      wcnt[NPAD];      // popcount(supm & valid) per row
    __shared__ unsigned vmask[32];       // valid bitmap
    __shared__ unsigned keepW[32];       // keep bitmap
    __shared__ int      gcnt;
    __shared__ int      thrBin;

    const int b   = blockIdx.x;
    const int tid = threadIdx.x;
    const float* pb = pred + (size_t)b * NROW * NCH;
    unsigned* supb = supw_g  + (size_t)b * NPAD * 32;
    float*    mg   = merged_g + (size_t)b * NPAD * 8;
    float*    ob   = out + (size_t)b * MAXDET * 6;

    for (int i = tid; i < MAXDET * 6; i += blockDim.x) ob[i] = 0.0f;

    // ---- pick score threshold from histogram (suffix count <= CAP) ----
    if (tid == 0) {
        unsigned sum = 0; int tb = 0;
        for (int i = NBINS - 1; i >= 0; --i) {
            unsigned h = hist[b * NBINS + i];
            if (sum + h > (unsigned)CAP) { tb = i + 1; break; }
            sum += h;
        }
        thrBin = tb;
        gcnt = 0;
    }
    for (int i = tid; i < NPAD; i += blockDim.x) wcnt[i] = 0;
    __syncthreads();
    const float thrLo = (float)thrBin * (1.0f / (float)NBINS);

    // ---- gather candidates above threshold into LDS ----
    for (int r = tid; r < NROW; r += blockDim.x) {
        const float* p = pb + (size_t)r * NCH;
        float obj = p[4];
        if (obj > CONF) {
            for (int c = 0; c < NC; ++c) {
                float s = p[5 + c] * obj;
                if (s > CONF && s >= thrLo) {
                    int pos = atomicAdd(&gcnt, 1);
                    if (pos < CAP) { U.s.sv[pos] = s; U.s.si[pos] = r * NC + c; }
                }
            }
        }
    }
    __syncthreads();
    int cnt = gcnt; if (cnt > CAP) cnt = CAP;
    for (int k = cnt + tid; k < CAP; k += blockDim.x) { U.s.sv[k] = -1e30f; U.s.si[k] = 0; }
    __syncthreads();

    // ---- bitonic sort descending over CAP (value,index) pairs ----
    for (unsigned kk = 2; kk <= CAP; kk <<= 1) {
        for (unsigned jj = kk >> 1; jj > 0; jj >>= 1) {
            for (int i = tid; i < CAP; i += blockDim.x) {
                unsigned ixj = (unsigned)i ^ jj;
                if (ixj > (unsigned)i) {
                    float va = U.s.sv[i], vb2 = U.s.sv[ixj];
                    bool up = ((unsigned)i & kk) == 0;
                    bool sw = up ? (va < vb2) : (va > vb2);
                    if (sw) {
                        U.s.sv[i] = vb2; U.s.sv[ixj] = va;
                        int ta = U.s.si[i]; U.s.si[i] = U.s.si[ixj]; U.s.si[ixj] = ta;
                    }
                }
            }
            __syncthreads();
        }
    }

    // ---- extract top NCAND: scores, boxes(xyxy), class ----
    for (int k = tid; k < NPAD; k += blockDim.x) {
        float v  = (k < NCAND) ? U.s.sv[k] : -1e30f;
        int   fi = (k < NCAND) ? U.s.si[k] : 0;
        topv[k] = v;
        tv16[k] = (v > CONF) ? (_Float16)v : (_Float16)0.0f;
        int bi = fi / NC, cj = fi - bi * NC;
        const float* p = pb + (size_t)bi * NCH;
        float cx = p[0], cy = p[1], w2 = p[2] * 0.5f, h2 = p[3] * 0.5f;
        bx[k][0] = cx - w2; bx[k][1] = cy - h2; bx[k][2] = cx + w2; bx[k][3] = cy + h2;
        clsA[k] = (float)cj;
    }
    __syncthreads();
    if (tid < 32) {
        unsigned w = 0;
        for (int t2 = 0; t2 < 32; ++t2)
            if (topv[tid * 32 + t2] > CONF) w |= 1u << t2;
        vmask[tid] = w;
        keepW[tid] = w;
    }
    __syncthreads();

    // ---- IoU matrix -> suppression bitmaps in global scratch ----
    for (int task = tid; task < NPAD * 32; task += blockDim.x) {
        int m = task >> 5, w = task & 31;
        float om = clsA[m] * MAX_WH;
        float mx0 = bx[m][0] + om, my0 = bx[m][1] + om;
        float mx1 = bx[m][2] + om, my1 = bx[m][3] + om;
        float am = (mx1 - mx0) * (my1 - my0);
        unsigned word = 0;
        #pragma unroll 4
        for (int t2 = 0; t2 < 32; ++t2) {
            int j = w * 32 + t2;
            float oj = clsA[j] * MAX_WH;
            float jx0 = bx[j][0] + oj, jy0 = bx[j][1] + oj;
            float jx1 = bx[j][2] + oj, jy1 = bx[j][3] + oj;
            float aj = (jx1 - jx0) * (jy1 - jy0);
            float lx = fmaxf(mx0, jx0), ly = fmaxf(my0, jy0);
            float rx = fminf(mx1, jx1), ry = fminf(my1, jy1);
            float iw = fmaxf(rx - lx, 0.0f), ih = fmaxf(ry - ly, 0.0f);
            float inter = iw * ih;
            float iou = inter / (am + aj - inter);
            if (iou > IOU_T) word |= 1u << t2;
        }
        supb[(size_t)m * 32 + w] = word;
        int pc = __popc(word & vmask[w]);
        if (pc) atomicAdd(&wcnt[m], pc);
    }
    __threadfence();
    __syncthreads();

    // ---- sequential greedy suppression: one wave, one bitmap word per lane ----
    if (tid < 32) {
        volatile unsigned* kp = keepW;
        const int w = tid;
        for (int i = 0; i < NCAND; ++i) {
            unsigned kw = kp[i >> 5];
            if ((kw >> (i & 31)) & 1u) {
                unsigned sup = supb[(size_t)i * 32 + w];
                int iw = i >> 5;
                unsigned mask;
                if (w < iw)      mask = 0u;
                else if (w > iw) mask = 0xFFFFFFFFu;
                else             mask = ((i & 31) == 31) ? 0u : (0xFFFFFFFFu << ((i & 31) + 1));
                unsigned clr = sup & mask;
                if (clr) kp[w] &= ~clr;
            }
        }
    }
    __syncthreads();
    if (tid == 0) {   // truncate keep to MAXDET by cumulative count
        int total = 0;
        for (int k = 0; k < NCAND; ++k) {
            if ((keepW[k >> 5] >> (k & 31)) & 1u) {
                ++total;
                if (total > MAXDET) keepW[k >> 5] &= ~(1u << (k & 31));
            }
        }
    }

    // ---- build transposed f16 B panel: BhT[n][k], cols = [x0 y0 x1 y1 1 0..] ----
    __syncthreads();
    for (int k = tid; k < NPAD; k += blockDim.x) {
        U.BhT[0][k] = (_Float16)bx[k][0];
        U.BhT[1][k] = (_Float16)bx[k][1];
        U.BhT[2][k] = (_Float16)bx[k][2];
        U.BhT[3][k] = (_Float16)bx[k][3];
        U.BhT[4][k] = (_Float16)1.0f;
        #pragma unroll
        for (int n = 5; n < 16; ++n) U.BhT[n][k] = (_Float16)0.0f;
    }
    __syncthreads();

    // ---- merge matmul on WMMA: merged[m][0..4] = sum_k (supm&valid)*s_k * B[k][n]
    {
        const int wave = tid >> 5;
        const int lane = tid & 31;
        const int hi = lane >> 4;
        const int lo = lane & 15;
        for (int t2 = wave; t2 < NPAD / 16; t2 += 8) {      // 64 M-tiles
            const int m0 = t2 * 16;
            const int mA = m0 + lo;                          // A row for this lane
            const unsigned* rowp = supb + (size_t)mA * 32;
            v8f acc = {};
            for (int kq = 0; kq < 8; ++kq) {                 // 4 bitmap words per b128
                uint4 aw4 = *(const uint4*)(rowp + kq * 4);
                unsigned aws[4] = {aw4.x, aw4.y, aw4.z, aw4.w};
                #pragma unroll
                for (int kk2 = 0; kk2 < 4; ++kk2) {
                    const int kbase = (kq * 4 + kk2) * 32;
                    const unsigned aw = aws[kk2];
                    // unconditional vector LDS loads (two 16B runs each)
                    h8 alo = *(const h8*)(tv16 + kbase + hi * 8);
                    h8 ahi = *(const h8*)(tv16 + kbase + 16 + hi * 8);
                    h8 blo = *(const h8*)(&U.BhT[lo][kbase + hi * 8]);
                    h8 bhi = *(const h8*)(&U.BhT[lo][kbase + 16 + hi * 8]);
                    v16h av, bv;
                    #pragma unroll
                    for (int e = 0; e < 8; ++e) {
                        _Float16 x0 = alo[e], x1 = ahi[e];   // loaded regardless
                        av[e]     = ((aw >> (hi * 8 + e)) & 1u)      ? x0 : (_Float16)0.0f;
                        av[8 + e] = ((aw >> (16 + hi * 8 + e)) & 1u) ? x1 : (_Float16)0.0f;
                        bv[e]     = blo[e];
                        bv[8 + e] = bhi[e];
                    }
                    acc = __builtin_amdgcn_wmma_f32_16x16x32_f16(
                        false, av, false, bv, (short)0, acc, false, false);
                }
            }
            if (lo < 5) {     // cols 0..3 = weighted box sums, col 4 = wsum
                #pragma unroll
                for (int r = 0; r < 8; ++r) {
                    int mo = m0 + r + hi * 8;     // C/D layout: VGPR r, lane half -> row
                    mg[(size_t)mo * 8 + lo] = acc[r];
                }
            }
        }
    }
    __syncthreads();

    // ---- compaction ranks (keep_final = keep & wcnt>1), then scatter rows ----
    if (tid == 0) {
        int rank = 0;
        for (int k = 0; k < NCAND; ++k) {
            bool kf = (((keepW[k >> 5] >> (k & 31)) & 1u) != 0) && (wcnt[k] > 1);
            U.dest[k] = kf ? rank : -1;
            if (kf) ++rank;
        }
        for (int k = NCAND; k < NPAD; ++k) U.dest[k] = -1;
    }
    __syncthreads();
    for (int k = tid; k < NCAND; k += blockDim.x) {
        int d = U.dest[k];
        if (d >= 0 && d < MAXDET) {
            float wsum = fmaxf(mg[(size_t)k * 8 + 4], 1e-12f);
            float inv = 1.0f / wsum;
            float* row = ob + (size_t)d * 6;
            row[0] = mg[(size_t)k * 8 + 0] * inv;
            row[1] = mg[(size_t)k * 8 + 1] * inv;
            row[2] = mg[(size_t)k * 8 + 2] * inv;
            row[3] = mg[(size_t)k * 8 + 3] * inv;
            row[4] = topv[k];
            row[5] = clsA[k];
        }
    }
}

extern "C" void kernel_launch(void* const* d_in, const int* in_sizes, int n_in,
                              void* d_out, int out_size, void* d_ws, size_t ws_size,
                              hipStream_t stream) {
    (void)in_sizes; (void)n_in; (void)out_size; (void)ws_size;
    const float* pred = (const float*)d_in[0];
    float* out = (float*)d_out;
    unsigned char* ws = (unsigned char*)d_ws;
    unsigned* hist   = (unsigned*)(ws + WS_HIST_OFF);
    unsigned* supw   = (unsigned*)(ws + WS_SUPW_OFF);
    float*    merged = (float*)   (ws + WS_MERG_OFF);

    nms_init<<<(NBATCH * NBINS + 255) / 256, 256, 0, stream>>>(hist);
    nms_hist<<<dim3(NBATCH, 32), 256, 0, stream>>>(pred, hist);
    nms_main<<<NBATCH, 256, 0, stream>>>(pred, hist, supw, merged, out);
}
